// GraphTransformerLayer_70042326663856
// MI455X (gfx1250) — compile-verified, matrix operand order checked by reading, not run on hardware
//
#include <hip/hip_runtime.h>
#include <hip/hip_fp16.h>

typedef __attribute__((ext_vector_type(16))) _Float16 v16h;
typedef __attribute__((ext_vector_type(8)))  _Float16 v8h;
typedef __attribute__((ext_vector_type(8)))  float    v8f;

#define D_   128
#define LDA  136   // halfs per LDS row (128 + 8 pad -> rows land 4 banks apart)
#define LDO  132   // floats per LDS row for LN staging

// Build a 16x32 f16 WMMA fragment (A or B^T) from an LDS row.
// ISA layout: lanes 0-15 (g=0): VGPR0-3 = K 0..7, VGPR4-7 = K 16..23
//             lanes 16-31 (g=1): VGPR0-3 = K 8..15, VGPR4-7 = K 24..31
__device__ __forceinline__ v16h load_frag(const _Float16* rowp, int k0, int g) {
  const v8h lo = *(const v8h*)(rowp + k0 + g * 8);
  const v8h hi = *(const v8h*)(rowp + k0 + 16 + g * 8);
  v16h r;
#pragma unroll
  for (int i = 0; i < 8; ++i) { r[i] = lo[i]; r[i + 8] = hi[i]; }
  return r;
}

__device__ __forceinline__ v8f wmma_f16(v16h a, v16h b, v8f c) {
  return __builtin_amdgcn_wmma_f32_16x16x32_f16(false, a, false, b, (short)0, c,
                                                false, false);
}

__device__ __forceinline__ void atomicMaxF(float* addr, float v) {
  if (v >= 0.0f) atomicMax((int*)addr, __float_as_int(v));
  else           atomicMin((unsigned int*)addr, __float_as_uint(v));
}

// ---------------------------------------------------------------------------
// Fused Q/K/V projection: stage x tile once, stream Wq/Wk/Wv through LDS.
// ---------------------------------------------------------------------------
__global__ __launch_bounds__(256) void qkv_gemm(
    const float* __restrict__ x,
    const float* __restrict__ Wq, const float* __restrict__ bq,
    const float* __restrict__ Wk, const float* __restrict__ bk,
    const float* __restrict__ Wv, const float* __restrict__ bv,
    float* __restrict__ Q, float* __restrict__ K, float* __restrict__ V,
    int n)
{
  __shared__ __align__(16) _Float16 As[64][LDA];
  __shared__ __align__(16) _Float16 Ws[128][LDA];   // transposed weight: [n][k]
  __shared__ float bs[128];
  const int tid = threadIdx.x;
  const int r0  = blockIdx.x * 64;

#pragma unroll
  for (int i = 0; i < 8; ++i) {                     // 64x128 f32 -> f16 LDS
    int idx = i * 256 + tid;
    int r = idx >> 5, c = (idx & 31) << 2;
    float4 v = make_float4(0.f, 0.f, 0.f, 0.f);
    if (r0 + r < n) v = *(const float4*)(x + (size_t)(r0 + r) * D_ + c);
    As[r][c + 0] = (_Float16)v.x; As[r][c + 1] = (_Float16)v.y;
    As[r][c + 2] = (_Float16)v.z; As[r][c + 3] = (_Float16)v.w;
  }

  const float* Wp[3] = {Wq, Wk, Wv};
  const float* bp[3] = {bq, bk, bv};
  float*       Op[3] = {Q, K, V};

  const int w = tid >> 5, lane = tid & 31;
  const int m0 = (w & 3) * 16, nh = (w >> 2) * 64;
  const int g = lane >> 4, lm = lane & 15;
  const int Mi = g * 8;

  for (int mat = 0; mat < 3; ++mat) {
    __syncthreads();                                // prior Ws readers done
#pragma unroll
    for (int i = 0; i < 16; ++i) {                  // W[k][n] -> Ws[n][k] f16
      int idx = i * 256 + tid;
      int k = idx >> 5, c = (idx & 31) << 2;
      float4 v = *(const float4*)(Wp[mat] + (size_t)k * D_ + c);
      Ws[c + 0][k] = (_Float16)v.x; Ws[c + 1][k] = (_Float16)v.y;
      Ws[c + 2][k] = (_Float16)v.z; Ws[c + 3][k] = (_Float16)v.w;
    }
    if (tid < 128) bs[tid] = bp[mat][tid];
    __syncthreads();

    v8f acc[4] = {};
#pragma unroll
    for (int kk = 0; kk < 4; ++kk) {
      v16h a = load_frag(&As[m0 + lm][0], kk * 32, g);
#pragma unroll
      for (int t = 0; t < 4; ++t) {
        v16h b = load_frag(&Ws[nh + t * 16 + lm][0], kk * 32, g);
        acc[t] = wmma_f16(a, b, acc[t]);
      }
    }
    float* Om = Op[mat];
#pragma unroll
    for (int t = 0; t < 4; ++t) {
      int col = nh + t * 16 + lm;
      float bv_ = bs[col];
#pragma unroll
      for (int j = 0; j < 8; ++j) {
        int row = r0 + m0 + Mi + j;
        if (row < n) Om[(size_t)row * D_ + col] = acc[t][j] + bv_;
      }
    }
  }
}

// ---------------------------------------------------------------------------
// Generic 128x128 GEMM with fused epilogue.
// EPI==1: out = gelu(A@W + b)
// EPI==2: out = layernorm(A@W + b + resid; gamma, beta)
// ---------------------------------------------------------------------------
template <int EPI>
__global__ __launch_bounds__(256) void gemm128(
    const float* __restrict__ A, const float* __restrict__ Wg,
    const float* __restrict__ bias, const float* __restrict__ resid,
    const float* __restrict__ gamma, const float* __restrict__ beta,
    float* __restrict__ out, int n)
{
  __shared__ __align__(16) _Float16 As[64][LDA];
  __shared__ __align__(16) unsigned char shraw[128 * LDA * 2];
  __shared__ float bs[128], gs[128], bb[128];
  _Float16 (*Ws)[LDA]  = (_Float16(*)[LDA])shraw;   // weight while computing
  float    (*Out)[LDO] = (float(*)[LDO])shraw;      // LN staging afterwards

  const int tid = threadIdx.x;
  const int r0  = blockIdx.x * 64;

#pragma unroll
  for (int i = 0; i < 8; ++i) {
    int idx = i * 256 + tid;
    int r = idx >> 5, c = (idx & 31) << 2;
    float4 v = make_float4(0.f, 0.f, 0.f, 0.f);
    if (r0 + r < n) v = *(const float4*)(A + (size_t)(r0 + r) * D_ + c);
    As[r][c + 0] = (_Float16)v.x; As[r][c + 1] = (_Float16)v.y;
    As[r][c + 2] = (_Float16)v.z; As[r][c + 3] = (_Float16)v.w;
  }
#pragma unroll
  for (int i = 0; i < 16; ++i) {
    int idx = i * 256 + tid;
    int k = idx >> 5, c = (idx & 31) << 2;
    float4 v = *(const float4*)(Wg + (size_t)k * D_ + c);
    Ws[c + 0][k] = (_Float16)v.x; Ws[c + 1][k] = (_Float16)v.y;
    Ws[c + 2][k] = (_Float16)v.z; Ws[c + 3][k] = (_Float16)v.w;
  }
  if (tid < 128) {
    bs[tid] = bias[tid];
    if (EPI == 2) { gs[tid] = gamma[tid]; bb[tid] = beta[tid]; }
  }
  __syncthreads();

  const int w = tid >> 5, lane = tid & 31;
  const int m0 = (w & 3) * 16, nh = (w >> 2) * 64;
  const int g = lane >> 4, lm = lane & 15;
  const int Mi = g * 8;

  v8f acc[4] = {};
#pragma unroll
  for (int kk = 0; kk < 4; ++kk) {
    v16h a = load_frag(&As[m0 + lm][0], kk * 32, g);
#pragma unroll
    for (int t = 0; t < 4; ++t) {
      v16h b = load_frag(&Ws[nh + t * 16 + lm][0], kk * 32, g);
      acc[t] = wmma_f16(a, b, acc[t]);
    }
  }

  if (EPI == 1) {
#pragma unroll
    for (int t = 0; t < 4; ++t) {
      int col = nh + t * 16 + lm;
      float bv_ = bs[col];
#pragma unroll
      for (int j = 0; j < 8; ++j) {
        int row = r0 + m0 + Mi + j;
        if (row < n) {
          float vv = acc[t][j] + bv_;
          out[(size_t)row * D_ + col] =
              0.5f * vv * (1.0f + erff(vv * 0.70710678118654752f));
        }
      }
    }
  } else {
    __syncthreads();                                // all Ws reads done
#pragma unroll
    for (int t = 0; t < 4; ++t) {
      int col = nh + t * 16 + lm;
#pragma unroll
      for (int j = 0; j < 8; ++j) Out[m0 + Mi + j][col] = acc[t][j];
    }
    __syncthreads();

    const int r = tid >> 2, q = tid & 3;            // 4 lanes per row
    const int grow = r0 + r;
    float4 vals[8];
    float s1 = 0.f, s2 = 0.f;
    if (grow < n) {
#pragma unroll
      for (int i = 0; i < 8; ++i) {
        int c = q * 32 + i * 4;
        float4 o  = *(float4*)&Out[r][c];
        float4 rv = *(const float4*)(resid + (size_t)grow * D_ + c);
        o.x += bs[c + 0] + rv.x; o.y += bs[c + 1] + rv.y;
        o.z += bs[c + 2] + rv.z; o.w += bs[c + 3] + rv.w;
        vals[i] = o;
        s1 += o.x + o.y + o.z + o.w;
        s2 += o.x * o.x + o.y * o.y + o.z * o.z + o.w * o.w;
      }
    }
    s1 += __shfl_xor(s1, 1); s1 += __shfl_xor(s1, 2);
    s2 += __shfl_xor(s2, 1); s2 += __shfl_xor(s2, 2);
    const float mean = s1 * (1.0f / 128.0f);
    const float var  = s2 * (1.0f / 128.0f) - mean * mean;
    const float rstd = rsqrtf(var + 1e-5f);
    if (grow < n) {
#pragma unroll
      for (int i = 0; i < 8; ++i) {
        int c = q * 32 + i * 4;
        float4 o = vals[i];
        o.x = (o.x - mean) * rstd * gs[c + 0] + bb[c + 0];
        o.y = (o.y - mean) * rstd * gs[c + 1] + bb[c + 1];
        o.z = (o.z - mean) * rstd * gs[c + 2] + bb[c + 2];
        o.w = (o.w - mean) * rstd * gs[c + 3] + bb[c + 3];
        *(float4*)(out + (size_t)grow * D_ + c) = o;
      }
    }
  }
}

// ---------------------------------------------------------------------------
// Edge phase: one wave per edge (32 lanes cover the 128-wide row).
// ---------------------------------------------------------------------------
__global__ __launch_bounds__(256) void edge_logits_kernel(
    const float* __restrict__ Q, const float* __restrict__ K,
    const int* __restrict__ ei, const int* __restrict__ et,
    const float* __restrict__ ebias, float* __restrict__ attn,
    float* __restrict__ mx, int E_)
{
  int idx = blockIdx.x * 256 + threadIdx.x;
  int e = idx >> 5;
  if (e >= E_) return;                  // wave-uniform
  int l = idx & 31;
  int src = ei[e], dst = ei[E_ + e];
  int d0 = l << 2;
  float4 qv = *(const float4*)(Q + (size_t)dst * D_ + d0);
  float4 kv = *(const float4*)(K + (size_t)src * D_ + d0);
  float p = qv.x * kv.x + qv.y * kv.y + qv.z * kv.z + qv.w * kv.w;
  p += __shfl_xor(p, 1);                // reduce 4 lanes (16 dims) per head
  p += __shfl_xor(p, 2);
  if ((l & 3) == 0) {
    int h = l >> 2;
    float a = p * 0.25f + ebias[et[e] * 8 + h];   // scale = 16^-0.5
    attn[(size_t)e * 8 + h] = a;
    atomicMaxF(mx + (size_t)dst * 8 + h, a);
  }
}

__global__ __launch_bounds__(256) void edge_expsum_kernel(
    const int* __restrict__ ei, float* __restrict__ attn,
    const float* __restrict__ mx, float* __restrict__ sm, int E_)
{
  int idx = blockIdx.x * 256 + threadIdx.x;
  if (idx >= E_ * 8) return;
  int e = idx >> 3, h = idx & 7;
  int dst = ei[E_ + e];
  float ex = expf(attn[idx] - mx[(size_t)dst * 8 + h]);
  attn[idx] = ex;
  atomicAdd(sm + (size_t)dst * 8 + h, ex);
}

__global__ __launch_bounds__(256) void edge_scatter_kernel(
    const int* __restrict__ ei, const float* __restrict__ attn,
    const float* __restrict__ sm, const float* __restrict__ V,
    float* __restrict__ agg, int E_)
{
  int idx = blockIdx.x * 256 + threadIdx.x;
  int e = idx >> 5;
  if (e >= E_) return;
  int l = idx & 31;
  int d0 = l << 2, h = l >> 2;
  int src = ei[e], dst = ei[E_ + e];
  float a = attn[(size_t)e * 8 + h] / (sm[(size_t)dst * 8 + h] + 1e-16f);
  float4 v = *(const float4*)(V + (size_t)src * D_ + d0);
  float* ap = agg + (size_t)dst * D_ + d0;
  atomicAdd(ap + 0, a * v.x);
  atomicAdd(ap + 1, a * v.y);
  atomicAdd(ap + 2, a * v.z);
  atomicAdd(ap + 3, a * v.w);
}

// ---------------------------------------------------------------------------
extern "C" void kernel_launch(void* const* d_in, const int* in_sizes, int n_in,
                              void* d_out, int out_size, void* d_ws,
                              size_t ws_size, hipStream_t stream)
{
  (void)n_in; (void)out_size; (void)ws_size;
  const float* x   = (const float*)d_in[0];
  const int*   ei  = (const int*)d_in[1];
  const int*   et  = (const int*)d_in[2];
  const float* Wq  = (const float*)d_in[3];
  const float* bq  = (const float*)d_in[4];
  const float* Wk  = (const float*)d_in[5];
  const float* bk  = (const float*)d_in[6];
  const float* Wv  = (const float*)d_in[7];
  const float* bv  = (const float*)d_in[8];
  const float* Wo  = (const float*)d_in[9];
  const float* bo  = (const float*)d_in[10];
  const float* ebv = (const float*)d_in[11];
  const float* W1  = (const float*)d_in[12];
  const float* b1  = (const float*)d_in[13];
  const float* W2  = (const float*)d_in[14];
  const float* b2  = (const float*)d_in[15];
  const float* g1  = (const float*)d_in[16];
  const float* be1 = (const float*)d_in[17];
  const float* g2  = (const float*)d_in[18];
  const float* be2 = (const float*)d_in[19];

  const int N = in_sizes[0] / D_;
  const int E = in_sizes[2];

  float* ws = (float*)d_ws;
  const size_t ND = (size_t)N * D_, NH = (size_t)N * 8, E8 = (size_t)E * 8;
  float* Q    = ws;
  float* Kf   = Q + ND;
  float* Vf   = Kf + ND;
  float* attn = Vf + ND;
  float* mx   = attn + E8;
  float* sm   = mx + NH;
  float* agg  = sm + NH;          // sm+agg contiguous -> single memset
  float* x1   = Q;                // Q dead after attention
  float* h1   = Kf;               // K dead after attention
  float* outp = (float*)d_out;

  hipMemsetAsync(mx, 0xFF, NH * sizeof(float), stream);       // -inf sentinel
  hipMemsetAsync(sm, 0, (NH + ND) * sizeof(float), stream);   // sm + agg

  const int gblk = (N + 63) / 64;
  const int eb32 = (int)(((size_t)E * 32 + 255) / 256);
  const int eb8  = (int)(((size_t)E * 8 + 255) / 256);

  qkv_gemm<<<gblk, 256, 0, stream>>>(x, Wq, bq, Wk, bk, Wv, bv, Q, Kf, Vf, N);
  edge_logits_kernel<<<eb32, 256, 0, stream>>>(Q, Kf, ei, et, ebv, attn, mx, E);
  edge_expsum_kernel<<<eb8, 256, 0, stream>>>(ei, attn, mx, sm, E);
  edge_scatter_kernel<<<eb32, 256, 0, stream>>>(ei, attn, sm, Vf, agg, E);
  gemm128<2><<<gblk, 256, 0, stream>>>(agg, Wo, bo, x, g1, be1, x1, N);
  gemm128<1><<<gblk, 256, 0, stream>>>(x1, W1, b1, nullptr, nullptr, nullptr, h1, N);
  gemm128<2><<<gblk, 256, 0, stream>>>(h1, W2, b2, x1, g2, be2, outp, N);
}
